// EncoderBlock_70050916598230
// MI455X (gfx1250) — matrix operand order repaired and checked by reading, not compiled, not run on hardware
//
#include <hip/hip_runtime.h>

typedef __attribute__((ext_vector_type(16))) __bf16 v16bf;
typedef __attribute__((ext_vector_type(8)))  float  v8f;
typedef __attribute__((ext_vector_type(4)))  unsigned int v4u;
typedef __attribute__((ext_vector_type(8)))  int v8i;
typedef __attribute__((ext_vector_type(4)))  int v4i;
typedef unsigned short u16;
typedef unsigned int   u32;
typedef unsigned long long u64;

#ifndef __has_builtin
#define __has_builtin(x) 0
#endif
#if __has_builtin(__builtin_amdgcn_tensor_load_to_lds) && __has_builtin(__builtin_amdgcn_s_wait_tensorcnt)
#define HAVE_TDM 1
#else
#define HAVE_TDM 0
#endif

__device__ __forceinline__ u16 f2bf(float f) {
    u32 u = __float_as_uint(f);
    u32 r = (u + 0x7FFFu + ((u >> 16) & 1u)) >> 16;   // round-to-nearest-even
    return (u16)r;
}

union FragBF { v16bf v; u32 u[8]; };

// Compiler-visible "LDS was written" barrier: the TDM intrinsic doesn't take the
// LDS pointer as an operand (it's hidden in the D# descriptor), so without this
// the compiler proves the shared array is never stored to and folds all the
// fragment ds_loads to undef. Escaping the pointer into a memory-clobber asm
// forces the loads to be materialized after the tensorcnt wait + barrier.
__device__ __forceinline__ void lds_publish(const void* p) {
    asm volatile("" : : "v"(p) : "memory");
}

// ---------------- fp32 -> bf16 convert ----------------
__global__ __launch_bounds__(256) void f32_to_bf16_kern(const float* __restrict__ in,
                                                        u16* __restrict__ out,
                                                        long long n) {
    long long i = (long long)blockIdx.x * blockDim.x + threadIdx.x;
    long long stride = (long long)gridDim.x * blockDim.x;
    for (; i < n; i += stride) out[i] = f2bf(in[i]);
}

// ---------------- fp32 [R,N] -> bf16 [N,R] transpose-convert ----------------
__global__ __launch_bounds__(256) void transpose_f32_bf16_kern(const float* __restrict__ in,
                                                               u16* __restrict__ out,
                                                               int R, int N) {
    __shared__ float tile[32][33];
    const int tx = threadIdx.x & 31, ty = threadIdx.x >> 5;     // 32x8
    const int r0 = blockIdx.y * 32, c0 = blockIdx.x * 32;
    #pragma unroll
    for (int j = 0; j < 4; ++j)
        tile[ty + j * 8][tx] = in[(long long)(r0 + ty + j * 8) * N + c0 + tx];
    __syncthreads();
    #pragma unroll
    for (int j = 0; j < 4; ++j)
        out[(long long)(c0 + ty + j * 8) * R + r0 + tx] = f2bf(tile[tx][ty + j * 8]);
}

// ---------------- TDM: async 128x32 bf16 tile -> padded LDS [128][36] ----------------
#if HAVE_TDM
__device__ __forceinline__ void tdm_load_tile(unsigned lds_byte,
                                              const u16* gtile,     // tile start (row-major)
                                              long long row_stride) // elements between rows
{
    u64 ga = (u64)(uintptr_t)gtile;
    v4u g0;
    g0[0] = 1u;                                            // count=1, user mode
    g0[1] = lds_byte;                                      // LDS byte address
    g0[2] = (u32)ga;                                       // global_addr[31:0]
    g0[3] = ((u32)(ga >> 32) & 0x1FFFFFFu) | (2u << 30);   // global_addr[56:32] | type=2
    // group1: data_size=1(2B), pad_enable, pad_interval=3 (16 DW = 64B row),
    // pad_amount=1 (2 DW = 4 halves) -> LDS row stride 36 halves.
    u32 w0 = (1u << 16) | (1u << 20) | (3u << 22) | (1u << 25);
    u64 d0 = (u64)row_stride;            // tensor_dim0 (OOB bound; tile always in-bounds)
    u32 d1 = 1u << 20;                   // tensor_dim1 (large)
    v8i g1;
    g1[0] = (int)w0;
    g1[1] = (int)((u32)(d0 & 0xFFFFu) << 16);                           // dim0[15:0]
    g1[2] = (int)(((u32)(d0 >> 16) & 0xFFFFu) | ((d1 & 0xFFFFu) << 16));// dim0[31:16]|dim1[15:0]
    g1[3] = (int)(((d1 >> 16) & 0xFFFFu) | (32u << 16));                // dim1[31:16]|tile_dim0=32
    g1[4] = (int)128u;                                                   // tile_dim1=128, tile_dim2=0
    g1[5] = (int)(u32)(row_stride & 0xFFFFFFFFull);                      // dim0_stride[31:0]
    g1[6] = (int)(u32)((row_stride >> 32) & 0xFFFFull);                  // dim0_stride[47:32]
    g1[7] = 0;
    v4i g2 = {1, 0, 0, 0};               // tensor_dim2=1 (safe), rest unused
    v4i g3 = {0, 0, 0, 0};
#if __clang_major__ >= 23
    v8i z8 = {0, 0, 0, 0, 0, 0, 0, 0};
    __builtin_amdgcn_tensor_load_to_lds(g0, g1, g2, g3, z8, 0);
#else
    __builtin_amdgcn_tensor_load_to_lds(g0, g1, g2, g3, 0);
#endif
}
#endif

// synchronous fallback staging: 128x32 bf16 tile -> LDS [128][36]
__device__ __forceinline__ void stage_tile(const u16* __restrict__ src, long long ld, int k0,
                                           u16* dst, int tid) {
    #pragma unroll
    for (int i = 0; i < 8; ++i) {
        int idx = tid + i * 256;             // 2048 u32 slots
        int row = idx >> 4;
        int cp  = idx & 15;
        u32 val = *(const u32*)(src + (long long)row * ld + k0 + cp * 2);
        *(u32*)(dst + row * 36 + cp * 2) = val;
    }
}

// one K-step: load fragments (ISA 7.12.2 layouts) + 8 WMMAs
__device__ __forceinline__ void mma_step(const u16* la, const u16* lb,
                                         int wm, int wn, int g, int hl,
                                         v8f acc[2][4]) {
    FragBF a[2], b[4];
    #pragma unroll
    for (int tm = 0; tm < 2; ++tm) {
        int mr = wm * 32 + tm * 16 + hl;
        #pragma unroll
        for (int v = 0; v < 8; ++v) {
            // A 16x32 bf16: VGPR v<4 -> K=2v(+8g) ; v>=4 -> K=2v+8(+8g)
            int ko = ((v < 4) ? (2 * v) : (2 * v + 8)) + 8 * g;
            a[tm].u[v] = *(const u32*)(la + mr * 36 + ko);
        }
    }
    #pragma unroll
    for (int tn = 0; tn < 4; ++tn) {
        int nc = wn * 64 + tn * 16 + hl;
        #pragma unroll
        for (int v = 0; v < 8; ++v)   // B 32x16: VGPR v<4 -> K=2v(+8g) ; v>=4 -> K=2v+8(+8g)
            b[tn].u[v] = *(const u32*)(lb + nc * 36 + ((v < 4) ? (2 * v) : (2 * v + 8)) + 8 * g);
    }
    #pragma unroll
    for (int tm = 0; tm < 2; ++tm)
        #pragma unroll
        for (int tn = 0; tn < 4; ++tn)
            acc[tm][tn] = __builtin_amdgcn_wmma_f32_16x16x32_bf16(
                false, a[tm].v, false, b[tn].v, (short)0, acc[tm][tn], false, false);
}

// ---------------- bf16 WMMA GEMM:  C = act(scale*(A @ B^T) + bias) ----------------
// A: [M,K] row-major (per batch, stride sA).  B: [N,ldb] row-major (operand B^T).
// Block tile 128x128, K-step 32, 8 waves, each wave 2x4 WMMA tiles.
// TDM path: double-buffered LDS, wave0 drives the Tensor Data Mover.
__global__ __launch_bounds__(256) void gemm_bf16_kern(
    const u16* __restrict__ A, const u16* __restrict__ B,
    const float* __restrict__ bias,
    float* __restrict__ Cf, u16* __restrict__ Cbf, int trans_bf,
    int M, int N, int K, long long ldb,
    long long sA, long long sB, long long sC,
    float scale, int relu)
{
    __shared__ u16 lds[4 * 4608];   // [buf0_A][buf0_B][buf1_A][buf1_B], rows padded to 36 halves

    const int tid  = threadIdx.x;
    const int wave = tid >> 5;
    const int lane = tid & 31;
    const int g    = lane >> 4;
    const int hl   = lane & 15;
    const int wm   = wave & 3;               // 4 row groups of 32
    const int wn   = wave >> 2;              // 2 col groups of 64

    const int m0 = blockIdx.y * 128;
    const int n0 = blockIdx.x * 128;
    const u16* Ap = A + (long long)blockIdx.z * sA + (long long)m0 * K;
    const u16* Bp = B + (long long)blockIdx.z * sB + (long long)n0 * ldb;

    v8f acc[2][4] = {};
    const int nsteps = K >> 5;

#if HAVE_TDM
    if (tid < 32) {
        tdm_load_tile(0,    Ap, K);
        tdm_load_tile(9216, Bp, ldb);
    }
    for (int step = 0; step < nsteps; ++step) {
        const int sel = step & 1;
        if (tid < 32) __builtin_amdgcn_s_wait_tensorcnt(0);
        __syncthreads();                      // publish buf[sel]; all done reading buf[sel^1]
        if (tid < 32 && step + 1 < nsteps) {  // prefetch next tiles while computing
            int k1 = (step + 1) << 5;
            tdm_load_tile((unsigned)((sel ^ 1) * 18432),        Ap + k1, K);
            tdm_load_tile((unsigned)((sel ^ 1) * 18432 + 9216), Bp + k1, ldb);
        }
        const u16* la = lds + sel * 9216;
        lds_publish(la);                      // compiler-side: LDS modified, reload fragments
        mma_step(la, la + 4608, wm, wn, g, hl, acc);
    }
#else
    for (int step = 0; step < nsteps; ++step) {
        int k0 = step << 5;
        stage_tile(Ap, K,   k0, lds,        tid);
        stage_tile(Bp, ldb, k0, lds + 4608, tid);
        __syncthreads();
        mma_step(lds, lds + 4608, wm, wn, g, hl, acc);
        __syncthreads();
    }
#endif

    // ---- epilogue: scale, bias, relu; fp32 and/or bf16 (optionally transposed) ----
    float* Cfp = Cf  ? Cf  + (long long)blockIdx.z * sC : nullptr;
    u16*   Cbp = Cbf ? Cbf + (long long)blockIdx.z * sC : nullptr;
    #pragma unroll
    for (int tm = 0; tm < 2; ++tm) {
        #pragma unroll
        for (int tn = 0; tn < 4; ++tn) {
            int nc = n0 + wn * 64 + tn * 16 + hl;
            float bb = bias ? bias[nc] : 0.0f;
            #pragma unroll
            for (int r = 0; r < 8; ++r) {
                int mr = m0 + wm * 32 + tm * 16 + r + 8 * g;   // C/D: VGPR r -> M=r (+8 hi lanes)
                float v = acc[tm][tn][r] * scale + bb;
                if (relu) v = fmaxf(v, 0.0f);
                if (Cfp) Cfp[(long long)mr * N + nc] = v;
                if (Cbp) {
                    if (trans_bf) Cbp[(long long)nc * M + mr] = f2bf(v);
                    else          Cbp[(long long)mr * N + nc] = f2bf(v);
                }
            }
        }
    }
}

// ---------------- row softmax (T=2048), bf16 out ----------------
__global__ __launch_bounds__(256) void softmax_bf16_kern(const float* __restrict__ S,
                                                         u16* __restrict__ P) {
    __shared__ float red[256];
    const int tid = threadIdx.x;
    const long long row = blockIdx.x;
    const float* Sr = S + row * 2048;
    u16* Pr = P + row * 2048;
    float loc[8];
    float m = -3.4e38f;
    #pragma unroll
    for (int i = 0; i < 8; ++i) { loc[i] = Sr[tid + i * 256]; m = fmaxf(m, loc[i]); }
    red[tid] = m; __syncthreads();
    for (int s = 128; s > 0; s >>= 1) { if (tid < s) red[tid] = fmaxf(red[tid], red[tid + s]); __syncthreads(); }
    m = red[0]; __syncthreads();
    float sum = 0.0f;
    #pragma unroll
    for (int i = 0; i < 8; ++i) { loc[i] = __expf(loc[i] - m); sum += loc[i]; }
    red[tid] = sum; __syncthreads();
    for (int s = 128; s > 0; s >>= 1) { if (tid < s) red[tid] += red[tid + s]; __syncthreads(); }
    float inv = 1.0f / red[0];
    #pragma unroll
    for (int i = 0; i < 8; ++i) Pr[tid + i * 256] = f2bf(loc[i] * inv);
}

// ---------------- fused residual add + LayerNorm (C=1024) ----------------
__global__ __launch_bounds__(256) void add_ln_kern(const float* __restrict__ X,
                                                   const float* __restrict__ Y,
                                                   const float* __restrict__ gamma,
                                                   const float* __restrict__ beta,
                                                   float* __restrict__ outF,
                                                   u16* __restrict__ outB) {
    __shared__ float red[256];
    const int tid = threadIdx.x;
    const long long row = blockIdx.x;
    const float* Xr = X + row * 1024;
    const float* Yr = Y + row * 1024;
    float v[4];
    float sum = 0.0f;
    #pragma unroll
    for (int i = 0; i < 4; ++i) { int c = tid + i * 256; v[i] = Xr[c] + Yr[c]; sum += v[i]; }
    red[tid] = sum; __syncthreads();
    for (int s = 128; s > 0; s >>= 1) { if (tid < s) red[tid] += red[tid + s]; __syncthreads(); }
    float mu = red[0] * (1.0f / 1024.0f); __syncthreads();
    float sq = 0.0f;
    #pragma unroll
    for (int i = 0; i < 4; ++i) { float d = v[i] - mu; sq += d * d; }
    red[tid] = sq; __syncthreads();
    for (int s = 128; s > 0; s >>= 1) { if (tid < s) red[tid] += red[tid + s]; __syncthreads(); }
    float rstd = rsqrtf(red[0] * (1.0f / 1024.0f) + 1e-5f);
    #pragma unroll
    for (int i = 0; i < 4; ++i) {
        int c = tid + i * 256;
        float o = (v[i] - mu) * rstd * gamma[c] + beta[c];
        if (outF) outF[row * 1024 + c] = o;
        if (outB) outB[row * 1024 + c] = f2bf(o);
    }
}

// ---------------- orchestration ----------------
extern "C" void kernel_launch(void* const* d_in, const int* in_sizes, int n_in,
                              void* d_out, int out_size, void* d_ws, size_t ws_size,
                              hipStream_t stream) {
    (void)in_sizes; (void)n_in; (void)out_size; (void)ws_size;
    const int Bb = 4, T = 2048, Cc = 1024, Hh = 2048;
    const long long BT = (long long)Bb * T;                 // 8192 rows

    const float* x   = (const float*)d_in[0];
    const float* Wq  = (const float*)d_in[1];
    const float* bq  = (const float*)d_in[2];
    const float* Wk  = (const float*)d_in[3];
    const float* bk  = (const float*)d_in[4];
    const float* Wv  = (const float*)d_in[5];
    const float* bv  = (const float*)d_in[6];
    const float* W1  = (const float*)d_in[7];
    const float* b1  = (const float*)d_in[8];
    const float* W2  = (const float*)d_in[9];
    const float* b2  = (const float*)d_in[10];
    const float* g1  = (const float*)d_in[11];
    const float* be1 = (const float*)d_in[12];
    const float* g2  = (const float*)d_in[13];
    const float* be2 = (const float*)d_in[14];
    float* out = (float*)d_out;

    char* wsp = (char*)d_ws;
    auto alloc = [&](long long bytes) {
        char* p = wsp;
        wsp += (bytes + 255) & ~255LL;
        return (void*)p;
    };
    u16*   x_bf   = (u16*)  alloc(BT * Cc * 2);
    u16*   wqT    = (u16*)  alloc((long long)Cc * Cc * 2);   // [C,C] transposed
    u16*   wkT    = (u16*)  alloc((long long)Cc * Cc * 2);
    u16*   wvT    = (u16*)  alloc((long long)Cc * Cc * 2);
    u16*   w1T    = (u16*)  alloc((long long)Cc * Hh * 2);   // [H,C]
    u16*   w2T    = (u16*)  alloc((long long)Hh * Cc * 2);   // [C,H]
    u16*   q_bf   = (u16*)  alloc(BT * Cc * 2);              // [8192,1024]
    u16*   k_bf   = (u16*)  alloc(BT * Cc * 2);              // [8192,1024]
    u16*   vT_bf  = (u16*)  alloc(BT * Cc * 2);              // [1024,8192] (transposed)
    float* scores = (float*)alloc((long long)Bb * T * T * 4);
    u16*   p_bf   = (u16*)  alloc((long long)Bb * T * T * 2);
    float* att    = (float*)alloc(BT * Cc * 4);
    float* h_f    = (float*)alloc(BT * Cc * 4);
    u16*   h_bf   = (u16*)  alloc(BT * Cc * 2);
    u16*   f1_bf  = (u16*)  alloc(BT * Hh * 2);
    float* f2_f   = (float*)alloc(BT * Cc * 4);

    // x -> bf16 (row-major A operand)
    f32_to_bf16_kern<<<dim3(4096), dim3(256), 0, stream>>>(x, x_bf, BT * Cc);

    // weights -> transposed bf16 (B^T operands: [N,K] row-major)
    auto tcvt = [&](const float* src, u16* dst, int R, int N) {
        transpose_f32_bf16_kern<<<dim3(N / 32, R / 32), dim3(256), 0, stream>>>(src, dst, R, N);
    };
    tcvt(Wq, wqT, Cc, Cc);
    tcvt(Wk, wkT, Cc, Cc);
    tcvt(Wv, wvT, Cc, Cc);
    tcvt(W1, w1T, Cc, Hh);
    tcvt(W2, w2T, Hh, Cc);

    const float inv_sqrt_c = 0.03125f;   // 1/sqrt(1024)

    // Q, K projections -> row-major bf16 ; V projection -> transposed bf16 [C, 8192]
    {
        dim3 grid(Cc / 128, (int)(BT / 128), 1);
        gemm_bf16_kern<<<grid, dim3(256), 0, stream>>>(
            x_bf, wqT, bq, nullptr, q_bf, 0, (int)BT, Cc, Cc, Cc, 0, 0, 0, 1.0f, 0);
        gemm_bf16_kern<<<grid, dim3(256), 0, stream>>>(
            x_bf, wkT, bk, nullptr, k_bf, 0, (int)BT, Cc, Cc, Cc, 0, 0, 0, 1.0f, 0);
        gemm_bf16_kern<<<grid, dim3(256), 0, stream>>>(
            x_bf, wvT, bv, nullptr, vT_bf, 1, (int)BT, Cc, Cc, Cc, 0, 0, 0, 1.0f, 0);
    }

    // scores = Q @ K^T / sqrt(C) per batch (K is already the [N,K] operand)
    {
        dim3 grid(T / 128, T / 128, Bb);
        gemm_bf16_kern<<<grid, dim3(256), 0, stream>>>(
            q_bf, k_bf, nullptr, scores, nullptr, 0, T, T, Cc, Cc,
            (long long)T * Cc, (long long)T * Cc, (long long)T * T, inv_sqrt_c, 0);
    }

    // softmax rows -> bf16 P
    softmax_bf16_kern<<<dim3((unsigned)(Bb * T)), dim3(256), 0, stream>>>(scores, p_bf);

    // attended = P @ V per batch ; B operand = V^T [C rows, 8192 cols], batch col offset T
    {
        dim3 grid(Cc / 128, T / 128, Bb);
        gemm_bf16_kern<<<grid, dim3(256), 0, stream>>>(
            p_bf, vT_bf, nullptr, att, nullptr, 0, T, Cc, T, BT,
            (long long)T * T, (long long)T, (long long)T * Cc, 1.0f, 0);
    }

    // h = LN(x + attended)
    add_ln_kern<<<dim3((unsigned)BT), dim3(256), 0, stream>>>(x, att, g1, be1, h_f, h_bf);

    // FFN1: relu(h @ W1 + b1) -> bf16
    {
        dim3 grid(Hh / 128, (int)(BT / 128), 1);
        gemm_bf16_kern<<<grid, dim3(256), 0, stream>>>(
            h_bf, w1T, b1, nullptr, f1_bf, 0, (int)BT, Hh, Cc, Cc, 0, 0, 0, 1.0f, 1);
    }
    // FFN2: f1 @ W2 + b2 -> fp32
    {
        dim3 grid(Cc / 128, (int)(BT / 128), 1);
        gemm_bf16_kern<<<grid, dim3(256), 0, stream>>>(
            f1_bf, w2T, b2, f2_f, nullptr, 0, (int)BT, Cc, Hh, Hh, 0, 0, 0, 1.0f, 0);
    }

    // out = LN(h + ffn)
    add_ln_kern<<<dim3((unsigned)BT), dim3(256), 0, stream>>>(h_f, f2_f, g2, be2, out, nullptr);
}
